// TimeAdaptiveAttention_987842478149
// MI455X (gfx1250) — matrix operand order repaired and checked by reading, not compiled, not run on hardware
//
#include <hip/hip_runtime.h>
#include <hip/hip_bf16.h>
#include <stdint.h>

#define BATCH 4
#define SEQ   2048
#define FDIM  512
#define HEADS 8
#define DHEAD 64
#define NELEM_PER_MAT (BATCH * HEADS * SEQ * DHEAD)  // 4194304 bf16 elems per Q/K/V
#define NX ((size_t)BATCH * SEQ * FDIM)              // 4194304 X elems
#define NW ((size_t)FDIM * FDIM)                     // 262144 elems per W

typedef __attribute__((ext_vector_type(16))) __bf16 v16bf;
typedef __attribute__((ext_vector_type(8)))  float  v8f;
typedef __attribute__((ext_vector_type(2)))  __bf16 v2bf;
typedef int v4i_vec __attribute__((vector_size(16)));   // matches async builtin param

#define AS1 __attribute__((address_space(1)))
#define AS3 __attribute__((address_space(3)))

#if defined(__has_builtin)
#  if __has_builtin(__builtin_amdgcn_global_load_async_to_lds_b128) && \
      __has_builtin(__builtin_amdgcn_s_wait_asynccnt)
#    define USE_ASYNC_LDS 1
#  endif
#endif

union Frag {
    v16bf v;
    uint4 q[2];
};

// packed f32x2 -> bf16x2 via native fptrunc (RTNE on gfx1250)
__device__ __forceinline__ unsigned int pack2(float a, float b) {
    union { v2bf v; unsigned int u; } c;
    c.v[0] = (__bf16)a;
    c.v[1] = (__bf16)b;
    return c.u;
}
__device__ __forceinline__ unsigned short f2bf(float x) {
    union { __bf16 h; unsigned short u; } c;
    c.h = (__bf16)x;
    return c.u;
}

// async (or fallback) 16-byte global->LDS copy
__device__ __forceinline__ void g2l_b128(const unsigned short* g, unsigned short* l) {
#if defined(USE_ASYNC_LDS)
    __builtin_amdgcn_global_load_async_to_lds_b128(
        (AS1 v4i_vec*)(const_cast<unsigned short*>(g)),
        (AS3 v4i_vec*)(l), 0, 0);
#else
    *(uint4*)l = *(const uint4*)g;
#endif
}
__device__ __forceinline__ void g2l_fence() {
#if defined(USE_ASYNC_LDS)
    __builtin_amdgcn_s_wait_asynccnt(0);
#endif
}

// ---------------------------------------------------------------------------
// Kernel 0a: one-shot f32 -> bf16 conversion of X and Wq/Wk/Wv into workspace.
// Each thread converts 8 contiguous floats (b128 in, b128 out).
// ---------------------------------------------------------------------------
__global__ __launch_bounds__(256) void cvt_bf16_kernel(
    const float* __restrict__ X,
    const float* __restrict__ Wq, const float* __restrict__ Wk, const float* __restrict__ Wv,
    unsigned short* __restrict__ Xbf, unsigned short* __restrict__ Wbf)
{
    const size_t i8 = ((size_t)blockIdx.x * 256 + threadIdx.x) * 8;
    const float* src;
    unsigned short* dst;
    if (i8 < NX) {
        src = X + i8;  dst = Xbf + i8;
    } else {
        const size_t w = i8 - NX;
        if      (w <     NW) { src = Wq +  w;          dst = Wbf + w; }
        else if (w < 2 * NW) { src = Wk + (w -     NW); dst = Wbf + w; }
        else if (w < 3 * NW) { src = Wv + (w - 2 * NW); dst = Wbf + w; }
        else return;
    }
    float4 f0 = ((const float4*)src)[0];
    float4 f1 = ((const float4*)src)[1];
    uint4 o;
    o.x = pack2(f0.x, f0.y); o.y = pack2(f0.z, f0.w);
    o.z = pack2(f1.x, f1.y); o.w = pack2(f1.z, f1.w);
    *(uint4*)dst = o;
}

// ---------------------------------------------------------------------------
// Kernel 0b: distance-bias table  zt[d] = (1 + e^vh/(1+e^{vh-ah*d})) / sqrt(D)
// ---------------------------------------------------------------------------
__global__ __launch_bounds__(256) void zt_kernel(
    const float* __restrict__ ahp, const float* __restrict__ vhp,
    float* __restrict__ zt)
{
    const int d = blockIdx.x * 256 + threadIdx.x;
    if (d < SEQ) {
        const float ah = ahp[0], vh = vhp[0];
        const float ev = __expf(vh);
        zt[d] = (1.f + ev / (1.f + __expf(vh - ah * (float)d))) * 0.125f;
    }
}

// ---------------------------------------------------------------------------
// Kernel 1: fused QKV projection from pre-converted bf16 X/W.
//   Q,K -> [B,H,S,D] bf16;  V -> TRANSPOSED [B,H,D,S] bf16.
// grid = (B*S/128, 24); 256 threads = 8 waves; pure async b128 staging.
// ---------------------------------------------------------------------------
#define LXS 40   // padded LDS row stride (halfwords) for 32-wide K slabs

__global__ __launch_bounds__(256) void qkv_proj_kernel(
    const unsigned short* __restrict__ Xbf, const unsigned short* __restrict__ Wbf,
    const float* __restrict__ bq, const float* __restrict__ bk, const float* __restrict__ bv,
    unsigned short* __restrict__ wsQ, unsigned short* __restrict__ wsK,
    unsigned short* __restrict__ wsV)
{
    __shared__ __align__(16) unsigned short lX[128 * LXS];
    __shared__ __align__(16) unsigned short lW[64 * LXS];

    const int tid  = threadIdx.x;
    const int lane = tid & 31;
    const int wave = tid >> 5;
    const int ln   = lane & 15;
    const int half = lane >> 4;

    const int rowTile = blockIdx.x;        // 128 rows of X (flat B*S)
    const int matid   = blockIdx.y >> 3;   // 0=Q 1=K 2=V
    const int head    = blockIdx.y & 7;

    const float* bias = (matid == 0) ? bq : (matid == 1 ? bk : bv);
    unsigned short* out = (matid == 0) ? wsQ : (matid == 1 ? wsK : wsV);
    const unsigned short* Wm = Wbf + (size_t)matid * NW;

    v8f acc[4];
#pragma unroll
    for (int n = 0; n < 4; ++n)
#pragma unroll
        for (int i = 0; i < 8; ++i) acc[n][i] = 0.f;

    const int srow  = tid >> 1;   // staged X row 0..127
    const int shalf = tid & 1;    // 16-col half of the 32-wide slab

    for (int k0 = 0; k0 < FDIM; k0 += 32) {
        // ---- stage X slab (128x32) and W slab (64x32): async b128, no VALU ----
        {
            const unsigned short* xs =
                Xbf + (size_t)(rowTile * 128 + srow) * FDIM + k0 + shalf * 16;
            unsigned short* xd = &lX[srow * LXS + shalf * 16];
            g2l_b128(xs,     xd);
            g2l_b128(xs + 8, xd + 8);
        }
        if (tid < 128) {
            const unsigned short* ws =
                Wm + (size_t)(head * 64 + (tid >> 1)) * FDIM + k0 + (tid & 1) * 16;
            unsigned short* wd = &lW[(tid >> 1) * LXS + (tid & 1) * 16];
            g2l_b128(ws,     wd);
            g2l_b128(ws + 8, wd + 8);
        }
        g2l_fence();
        __syncthreads();

        // ---- A fragment (16x32): lane ln = row, K-chunks {kb..kb+7, kb+16..kb+23} ----
        Frag a;
        {
            const int r  = wave * 16 + ln;
            const int kb = half * 8;
            a.q[0] = *(const uint4*)&lX[r * LXS + kb];
            a.q[1] = *(const uint4*)&lX[r * LXS + kb + 16];
        }
        // ---- B fragments (32x16): lane ln = col, K-chunks {kb..kb+7, kb+16..kb+23} ----
#pragma unroll
        for (int n = 0; n < 4; ++n) {
            Frag bf;
            const int wr = n * 16 + ln;
            bf.q[0] = *(const uint4*)&lW[wr * LXS + half * 8];
            bf.q[1] = *(const uint4*)&lW[wr * LXS + half * 8 + 16];
            acc[n] = __builtin_amdgcn_wmma_f32_16x16x32_bf16(
                false, a.v, false, bf.v, (short)0, acc[n], false, false);
        }
        __syncthreads();
    }

    // ---- epilogue ----
    if (matid == 2) {
        // V: transposed layout [B,H,D,S]; rows i,i+1 -> s,s+1 => packed b32 stores
#pragma unroll
        for (int n = 0; n < 4; ++n) {
            const int d = n * 16 + ln;
            const float bvv = bias[head * 64 + d];
#pragma unroll
            for (int i = 0; i < 8; i += 2) {
                const int m = i + half * 8;
                const int r = rowTile * 128 + wave * 16 + m;
                const int b = r / SEQ, s = r % SEQ;
                const unsigned int pr = pack2(acc[n][i] + bvv, acc[n][i + 1] + bvv);
                *(unsigned int*)&out[(((size_t)b * HEADS + head) * DHEAD + d) * SEQ + s] = pr;
            }
        }
    } else {
        // Q,K: [B,H,S,D]
#pragma unroll
        for (int n = 0; n < 4; ++n) {
            const int d = n * 16 + ln;
            const float bvv = bias[head * 64 + d];
#pragma unroll
            for (int i = 0; i < 8; ++i) {
                const int m = i + half * 8;
                const int r = rowTile * 128 + wave * 16 + m;
                const int b = r / SEQ, s = r % SEQ;
                out[(((size_t)b * HEADS + head) * SEQ + s) * DHEAD + d] =
                    f2bf(acc[n][i] + bvv);
            }
        }
    }
}

// ---------------------------------------------------------------------------
// Kernel 2: flash attention, bias via LDS-resident z_hat table.
// grid = (S/128, H, B); 256 threads = 8 waves, each wave owns 16 query rows.
// Softmax pass covers the whole 64-key tile (fewer reductions / rescales).
// ---------------------------------------------------------------------------
#define LKS 72   // padded row stride (halfwords) for 64-wide tiles
#define LPS 72   // padded row stride for the 16x64 P tile

__global__ __launch_bounds__(256) void flash_attn_kernel(
    const unsigned short* __restrict__ wsQ,
    const unsigned short* __restrict__ wsK,
    const unsigned short* __restrict__ wsVt,
    const float* __restrict__ zt,
    float* __restrict__ out)
{
    __shared__ __align__(16) unsigned short lK[64 * LKS];    // [key][d]
    __shared__ __align__(16) unsigned short lVt[64 * LKS];   // [d][key]
    __shared__ __align__(16) unsigned short lP[8][16 * LPS]; // per-wave P [16][64]
    __shared__ __align__(16) float lZ[SEQ];                  // z_hat/sqrt(D) table

    const int tid  = threadIdx.x;
    const int lane = tid & 31;
    const int wave = tid >> 5;
    const int ln   = lane & 15;
    const int half = lane >> 4;

    const int qTile = blockIdx.x;
    const int h     = blockIdx.y;
    const int b     = blockIdx.z;

    // stage the full distance table once (2048 floats, 8 per thread)
    ((float4*)lZ)[tid * 2]     = ((const float4*)zt)[tid * 2];
    ((float4*)lZ)[tid * 2 + 1] = ((const float4*)zt)[tid * 2 + 1];

    const size_t bhBase = ((size_t)b * HEADS + h) * SEQ * DHEAD;  // same for [H,D,S]
    const int qRow0 = qTile * 128 + wave * 16;

    // Q A-fragments for d 0..31 and 32..63 (live in registers for whole kernel)
    Frag qa[2];
    {
        const unsigned short* qp = wsQ + bhBase + (size_t)(qRow0 + ln) * DHEAD;
        const int kb = half * 8;
#pragma unroll
        for (int dh = 0; dh < 2; ++dh) {
            qa[dh].q[0] = *(const uint4*)(qp + dh * 32 + kb);
            qa[dh].q[1] = *(const uint4*)(qp + dh * 32 + kb + 16);
        }
    }

    v8f o[4];
#pragma unroll
    for (int n = 0; n < 4; ++n)
#pragma unroll
        for (int i = 0; i < 8; ++i) o[n][i] = 0.f;

    float mrow[8], lrow[8];
#pragma unroll
    for (int i = 0; i < 8; ++i) { mrow[i] = -1e30f; lrow[i] = 0.f; }

    const int srow = tid >> 2;        // 0..63 (key row for K, d row for Vt)
    const int sq   = tid & 3;         // 16-element quarter

    for (int kk0 = 0; kk0 < SEQ; kk0 += 64) {
        __syncthreads();   // previous tile fully consumed (also covers lZ staging)
        // ---- stage K [key][d] and Vt [d][key] via async LDS loads ----
        {
            const unsigned short* ks  = wsK  + bhBase + (size_t)(kk0 + srow) * DHEAD + sq * 16;
            const unsigned short* vts = wsVt + bhBase + (size_t)srow * SEQ + kk0 + sq * 16;
            unsigned short* kd = &lK[srow * LKS + sq * 16];
            unsigned short* vd = &lVt[srow * LKS + sq * 16];
            g2l_b128(ks,      kd);
            g2l_b128(ks + 8,  kd + 8);
            g2l_b128(vts,     vd);
            g2l_b128(vts + 8, vd + 8);
            g2l_fence();
        }
        __syncthreads();

        // ---- S = Q K^T over all 64 keys : 8 WMMA into 4 score tiles ----
        v8f sc[4];
#pragma unroll
        for (int n = 0; n < 4; ++n)
#pragma unroll
            for (int i = 0; i < 8; ++i) sc[n][i] = 0.f;
#pragma unroll
        for (int dh = 0; dh < 2; ++dh) {
#pragma unroll
            for (int sub = 0; sub < 4; ++sub) {
                Frag bf;
                const int base = (sub * 16 + ln) * LKS + dh * 32 + half * 8;
                bf.q[0] = *(const uint4*)&lK[base];
                bf.q[1] = *(const uint4*)&lK[base + 16];
                sc[sub] = __builtin_amdgcn_wmma_f32_16x16x32_bf16(
                    false, qa[dh].v, false, bf.v, (short)0, sc[sub], false, false);
            }
        }

        // ---- bias via table lookup + single online-softmax pass over 64 keys ----
        unsigned short* lPw = &lP[wave][0];
#pragma unroll
        for (int i = 0; i < 8; ++i) {
            const int m    = i + half * 8;
            const int qi   = qRow0 + m;
            const int bofs = qi - kk0 - ln;   // d = |bofs - sub*16|
            float vv[4];
#pragma unroll
            for (int sub = 0; sub < 4; ++sub) {
                int d = bofs - sub * 16;
                d = (d < 0) ? -d : d;
                vv[sub] = sc[sub][i] * lZ[d];
            }
            float t = fmaxf(fmaxf(vv[0], vv[1]), fmaxf(vv[2], vv[3]));
#pragma unroll
            for (int msk = 8; msk; msk >>= 1) t = fmaxf(t, __shfl_xor(t, msk, 32));
            const float mnew  = fmaxf(mrow[i], t);
            const float alpha = __expf(mrow[i] - mnew);
            float sum = 0.f;
#pragma unroll
            for (int sub = 0; sub < 4; ++sub) {
                const float p = __expf(vv[sub] - mnew);
                sum += p;
                lPw[m * LPS + sub * 16 + ln] = f2bf(p);
            }
#pragma unroll
            for (int msk = 8; msk; msk >>= 1) sum += __shfl_xor(sum, msk, 32);
            lrow[i] = lrow[i] * alpha + sum;
            mrow[i] = mnew;
#pragma unroll
            for (int nd = 0; nd < 4; ++nd) o[nd][i] *= alpha;
        }

        // ---- P A-fragments (relayout via per-wave LDS; DS in-order per wave) ----
        Frag pa[2];
        {
            const int kb = half * 8;
#pragma unroll
            for (int c = 0; c < 2; ++c) {
                pa[c].q[0] = *(const uint4*)&lPw[ln * LPS + c * 32 + kb];
                pa[c].q[1] = *(const uint4*)&lPw[ln * LPS + c * 32 + kb + 16];
            }
        }

        // ---- O += P V : 8 WMMA from the [d][key] tile ----
#pragma unroll
        for (int c = 0; c < 2; ++c) {
#pragma unroll
            for (int nd = 0; nd < 4; ++nd) {
                Frag bv;
                const int base = (nd * 16 + ln) * LKS + c * 32 + half * 8;
                bv.q[0] = *(const uint4*)&lVt[base];
                bv.q[1] = *(const uint4*)&lVt[base + 16];
                o[nd] = __builtin_amdgcn_wmma_f32_16x16x32_bf16(
                    false, pa[c].v, false, bv.v, (short)0, o[nd], false, false);
            }
        }
    }

    // ---- normalize and store f32 output [B,S,H*D] ----
#pragma unroll
    for (int i = 0; i < 8; ++i) {
        const float inv = 1.f / lrow[i];
        const int   m   = i + half * 8;
        const int   s   = qRow0 + m;
        float* orow = out + ((size_t)b * SEQ + s) * FDIM + h * DHEAD;
#pragma unroll
        for (int nd = 0; nd < 4; ++nd)
            orow[nd * 16 + ln] = o[nd][i] * inv;
    }
}

// ---------------------------------------------------------------------------
extern "C" void kernel_launch(void* const* d_in, const int* in_sizes, int n_in,
                              void* d_out, int out_size, void* d_ws, size_t ws_size,
                              hipStream_t stream) {
    const float* X  = (const float*)d_in[0];
    const float* Wq = (const float*)d_in[1];
    const float* bq = (const float*)d_in[2];
    const float* Wk = (const float*)d_in[3];
    const float* bk = (const float*)d_in[4];
    const float* Wv = (const float*)d_in[5];
    const float* bv = (const float*)d_in[6];
    const float* ah = (const float*)d_in[7];
    const float* vh = (const float*)d_in[8];
    float* out = (float*)d_out;

    // workspace carve-up (all 16B-aligned)
    unsigned short* wsQ  = (unsigned short*)d_ws;
    unsigned short* wsK  = wsQ + NELEM_PER_MAT;
    unsigned short* wsVt = wsK + NELEM_PER_MAT;
    unsigned short* Xbf  = wsVt + NELEM_PER_MAT;
    unsigned short* Wbf  = Xbf + NX;
    float*          zt   = (float*)(Wbf + 3 * NW);

    // 0a: convert X + W to bf16 (one shot);  0b: distance-bias table
    const size_t totalCvt = NX + 3 * NW;
    cvt_bf16_kernel<<<(unsigned)(totalCvt / 8 / 256), 256, 0, stream>>>(
        X, Wq, Wk, Wv, Xbf, Wbf);
    zt_kernel<<<SEQ / 256, 256, 0, stream>>>(ah, vh, zt);

    // 1: QKV projection
    dim3 g1(BATCH * SEQ / 128, 24);
    qkv_proj_kernel<<<g1, 256, 0, stream>>>(Xbf, Wbf, bq, bk, bv, wsQ, wsK, wsVt);

    // 2: flash attention
    dim3 g2(SEQ / 128, HEADS, BATCH);
    flash_attn_kernel<<<g2, 256, 0, stream>>>(wsQ, wsK, wsVt, zt, out);
}
